// TensorProductConvLayer_40200893891319
// MI455X (gfx1250) — matrix operand order repaired and checked by reading, not compiled, not run on hardware
//
#include <hip/hip_runtime.h>

typedef __attribute__((ext_vector_type(16))) _Float16 v16h;
typedef __attribute__((ext_vector_type(8)))  float    v8f;

#define INV3  0.5773502691896258f   /* 1/sqrt(3)  */
#define NTP   0.22360679774997896f  /* 1/sqrt(20) */
#define NATT  0.31622776601683794f  /* 1/sqrt(10) */

// ---------------- LDS layout (bytes), 32B aligned blocks ----------------
#define L_W1K  0
#define L_W2K  (L_W1K + 32*64*2)        /* 4096   */
#define L_W3K  (L_W2K + 64*64*2)        /* 12288  */
#define L_W1V  (L_W3K + 64*208*2)       /* 38912  */
#define L_W2V  (L_W1V + 32*64*2)        /* 43008  */
#define L_W3V  (L_W2V + 64*64*2)        /* 51200  */
#define L_B1K  (L_W3V + 64*400*2)       /* 102400 */
#define L_B2K  (L_B1K + 64*4)
#define L_B3K  (L_B2K + 64*4)
#define L_B1V  (L_B3K + 208*4)
#define L_B2V  (L_B1V + 64*4)
#define L_B3V  (L_B2V + 64*4)
#define L_WD0  (L_B3V + 400*4)
#define L_WD1  (L_WD0 + 64*4)
#define L_HBUF (L_WD1 + 32)             /* pad -> 32B aligned           */
#define L_WBUF (L_HBUF + 4*1024*2)      /* 4 waves * 2 A-frags * 512 h  */
#define L_TOTAL (L_WBUF + 4*16*400*4)   /* 4 waves * 16x400 f32         */

static __device__ __forceinline__ void ldsfence() {
  asm volatile("s_wait_dscnt 0" ::: "memory");
}

static __device__ __forceinline__ v8f wmma16(v16h a, v16h b, v8f c) {
  return __builtin_amdgcn_wmma_f32_16x16x32_f16(false, a, false, b, (short)0, c,
                                                false, false);
}

// Contiguous fragment load: fragment = 32 lanes x 16 f16 (32B per lane).
static __device__ __forceinline__ v16h ldfrag(const _Float16* base, int f, int lane) {
  return *((const v16h*)base + (size_t)f * 32 + lane);
}

// Store C fragment (16x16 f32) into A-fragment order (f16, with ReLU).
// A layout (16-bit, 16x32): lane group hh = (k>>3)&1, elem j = ((k>>4)<<3)|(k&7).
static __device__ __forceinline__ void st_afrag(_Float16* abuf, int nb, v8f c, int lane) {
  int n = lane & 15, mb = (lane >> 4) * 8;
  int col = nb * 16 + n;                  // col = K index of next layer
  int kb = col >> 5, kk = col & 31;
  int hh = (kk >> 3) & 1;
  int j  = ((kk >> 4) << 3) | (kk & 7);
  _Float16* p = abuf + ((kb * 32 + hh * 16 + mb) * 16 + j);
#pragma unroll
  for (int r = 0; r < 8; ++r) p[r * 16] = (_Float16)fmaxf(c[r], 0.0f);
}

// Convert f32 row-major KxNcols weight into pre-swizzled B fragments in LDS.
static __device__ __forceinline__
void stage_bfrags(const float* __restrict__ W, _Float16* dst, int K, int ncs, int ncp, int tid) {
  int total = K * ncp;
  for (int i = tid; i < total; i += 128) {
    int k = i / ncp, col = i - k * ncp;
    float v = (col < ncs) ? W[k * ncs + col] : 0.0f;
    int kr = k & 31, kb = k >> 5;
    int nb = col >> 4, n = col & 15;
    int lane = (((kr >> 3) & 1) << 4) | n; // B layout: (k>>3)&1 picks lane group
    int j = ((kr >> 4) << 3) | (kr & 7);  // elem j = ((k>>4)<<3)|(k&7), matches A
    int f = nb * (K >> 5) + kb;
    dst[((size_t)f * 32 + lane) * 16 + j] = (_Float16)v;
  }
}

// One 16-row tile through a 3-layer MLP: 32->64 relu ->64 relu ->ncols.
static __device__ __forceinline__
void mlp_tile(v16h xa,
              const _Float16* W1, const float* B1,
              const _Float16* W2, const float* B2,
              const _Float16* W3, const float* B3,
              int nblk3, int ld3,
              _Float16* abuf, float* wbuf, int lane) {
  const int n = lane & 15, mb = (lane >> 4) * 8;
  // layer 1: (16x32)@(32x64)
#pragma unroll
  for (int nb = 0; nb < 4; ++nb) {
    float b = B1[nb * 16 + n];
    v8f c;
#pragma unroll
    for (int r = 0; r < 8; ++r) c[r] = b;
    c = wmma16(xa, ldfrag(W1, nb, lane), c);
    st_afrag(abuf, nb, c, lane);
  }
  ldsfence();
  v16h a0 = ldfrag(abuf, 0, lane);
  v16h a1 = ldfrag(abuf, 1, lane);
  // layer 2: (16x64)@(64x64)
#pragma unroll
  for (int nb = 0; nb < 4; ++nb) {
    float b = B2[nb * 16 + n];
    v8f c;
#pragma unroll
    for (int r = 0; r < 8; ++r) c[r] = b;
    c = wmma16(a0, ldfrag(W2, nb * 2 + 0, lane), c);
    c = wmma16(a1, ldfrag(W2, nb * 2 + 1, lane), c);
    st_afrag(abuf, nb, c, lane);
  }
  ldsfence();
  a0 = ldfrag(abuf, 0, lane);
  a1 = ldfrag(abuf, 1, lane);
  // layer 3: (16x64)@(64xncols) -> row-major f32 in wbuf
  for (int nb = 0; nb < nblk3; ++nb) {
    float b = B3[nb * 16 + n];
    v8f c;
#pragma unroll
    for (int r = 0; r < 8; ++r) c[r] = b;
    c = wmma16(a0, ldfrag(W3, nb * 2 + 0, lane), c);
    c = wmma16(a1, ldfrag(W3, nb * 2 + 1, lane), c);
#pragma unroll
    for (int r = 0; r < 8; ++r) wbuf[(mb + r) * ld3 + nb * 16 + n] = c[r];
  }
  ldsfence();
}

// ------------------------- Kernel 1: node linears -------------------------
__global__ void tpconv_node_kernel(const float* __restrict__ na,
                                   const float* __restrict__ Win0, const float* __restrict__ Win1,
                                   const float* __restrict__ Wq0,  const float* __restrict__ Wq1,
                                   float* __restrict__ xw, float* __restrict__ qw, int N) {
  int nd = blockIdx.x * blockDim.x + threadIdx.x;
  if (nd >= N) return;
  const float* r = na + (size_t)nd * 56;
  float x0[16], x1[12];
#pragma unroll
  for (int o = 0; o < 16; ++o) {
    float s = 0.f;
#pragma unroll
    for (int i = 0; i < 32; ++i) s += r[i] * Win0[i * 16 + o];
    x0[o] = s * 0.17677669529663687f;   // 1/sqrt(32)
  }
#pragma unroll
  for (int o = 0; o < 4; ++o)
#pragma unroll
    for (int c = 0; c < 3; ++c) {
      float s = 0.f;
#pragma unroll
      for (int m = 0; m < 8; ++m) s += r[32 + m * 3 + c] * Win1[m * 4 + o];
      x1[o * 3 + c] = s * 0.3535533905932738f;  // 1/sqrt(8)
    }
  float* xo = xw + (size_t)nd * 28;
#pragma unroll
  for (int i = 0; i < 16; ++i) xo[i] = x0[i];
#pragma unroll
  for (int i = 0; i < 12; ++i) xo[16 + i] = x1[i];
  float* qo = qw + (size_t)nd * 14;
#pragma unroll
  for (int j = 0; j < 8; ++j) {
    float s = 0.f;
#pragma unroll
    for (int o = 0; o < 16; ++o) s += x0[o] * Wq0[o * 8 + j];
    qo[j] = s * 0.25f;                  // 1/sqrt(16)
  }
#pragma unroll
  for (int j = 0; j < 2; ++j)
#pragma unroll
    for (int c = 0; c < 3; ++c) {
      float s = 0.f;
#pragma unroll
      for (int o = 0; o < 4; ++o) s += x1[o * 3 + c] * Wq1[o * 2 + j];
      qo[8 + j * 3 + c] = s * 0.5f;     // 1/sqrt(4)
    }
}

// ------------------- Kernel 2: fused edge MLPs + tp_fc --------------------
__global__ __launch_bounds__(128)
void tpconv_edge_kernel(const float* __restrict__ edge_attr, const float* __restrict__ edge_sh,
                        const int* __restrict__ eidx,
                        const float* __restrict__ xw, const float* __restrict__ qw,
                        const float* fkW1, const float* fkb1, const float* fkW2, const float* fkb2,
                        const float* fkW3, const float* fkb3,
                        const float* fcW1, const float* fcb1, const float* fcW2, const float* fcb2,
                        const float* fcW3, const float* fcb3,
                        const float* Wd0, const float* Wd1,
                        float* __restrict__ alog, float* __restrict__ vout, int E) {
  extern __shared__ char smem[];
  _Float16* w1k = (_Float16*)(smem + L_W1K);
  _Float16* w2k = (_Float16*)(smem + L_W2K);
  _Float16* w3k = (_Float16*)(smem + L_W3K);
  _Float16* w1v = (_Float16*)(smem + L_W1V);
  _Float16* w2v = (_Float16*)(smem + L_W2V);
  _Float16* w3v = (_Float16*)(smem + L_W3V);
  float* b1k = (float*)(smem + L_B1K);  float* b2k = (float*)(smem + L_B2K);
  float* b3k = (float*)(smem + L_B3K);  float* b1v = (float*)(smem + L_B1V);
  float* b2v = (float*)(smem + L_B2V);  float* b3v = (float*)(smem + L_B3V);
  float* wd0s = (float*)(smem + L_WD0); float* wd1s = (float*)(smem + L_WD1);

  const int tid = threadIdx.x;
  // Stage all weights once per block (f32 -> f16, pre-swizzled B fragments).
  stage_bfrags(fkW1, w1k, 32, 64, 64, tid);
  stage_bfrags(fcW1, w1v, 32, 64, 64, tid);
  stage_bfrags(fkW2, w2k, 64, 64, 64, tid);
  stage_bfrags(fcW2, w2v, 64, 64, 64, tid);
  stage_bfrags(fkW3, w3k, 64, 200, 208, tid);
  stage_bfrags(fcW3, w3v, 64, 400, 400, tid);
  for (int i = tid; i < 64; i += 128) {
    b1k[i] = fkb1[i]; b2k[i] = fkb2[i];
    b1v[i] = fcb1[i]; b2v[i] = fcb2[i];
    wd0s[i] = Wd0[i];
  }
  for (int i = tid; i < 208; i += 128) b3k[i] = (i < 200) ? fkb3[i] : 0.f;
  for (int i = tid; i < 400; i += 128) b3v[i] = fcb3[i];
  if (tid < 4) wd1s[tid] = Wd1[tid];
  __syncthreads();

  const int wid = tid >> 5, lane = tid & 31;
  const int m = lane & 15, hh = lane >> 4;
  _Float16* abuf = (_Float16*)(smem + L_HBUF) + (size_t)wid * 1024;
  float*    wbuf = (float*)(smem + L_WBUF) + (size_t)wid * 16 * 400;

  const int ntiles = (E + 15) >> 4;
  for (int tile = blockIdx.x * 4 + wid; tile < ntiles; tile += gridDim.x * 4) {
    int e = tile * 16 + m;
    bool ev = (e < E);
    int es = ev ? e : 0;

    // Build A fragment of edge_attr (16 edges x 32 feats) straight from HBM.
    const float* row = edge_attr + (size_t)es * 32;
    v16h xa;
#pragma unroll
    for (int j = 0; j < 8; ++j) xa[j] = ev ? (_Float16)row[8 * hh + j] : (_Float16)0.f;
#pragma unroll
    for (int j = 0; j < 8; ++j) xa[8 + j] = ev ? (_Float16)row[16 + 8 * hh + j] : (_Float16)0.f;

    // Per-edge operands (each lane handles edge m; halves split v work).
    int s = eidx[es], d = eidx[(size_t)E + es];
    float xs[28];
#pragma unroll
    for (int a = 0; a < 28; ++a) xs[a] = xw[(size_t)s * 28 + a];
    float sh0 = edge_sh[(size_t)es * 4];
    float sh1[3];
#pragma unroll
    for (int c = 0; c < 3; ++c) sh1[c] = edge_sh[(size_t)es * 4 + 1 + c];
    float qd[14];
#pragma unroll
    for (int i = 0; i < 14; ++i) qd[i] = qw[(size_t)d * 14 + i];

    // ================= K path: MLP -> tp_fc -> attention logit ============
    mlp_tile(xa, w1k, b1k, w2k, b2k, w3k, b3k, 13, 208, abuf, wbuf, lane);
    const float* wk = wbuf + m * 208;   // [w00 16x8 |128| w01 16x2 |160| w10 4x2 |168| w11 4x8]
    float k0v[8], k1v[6];
#pragma unroll
    for (int b = 0; b < 8; ++b) {
      float s0 = 0.f, t = 0.f;
#pragma unroll
      for (int a = 0; a < 16; ++a) s0 += xs[a] * wk[a * 8 + b];
#pragma unroll
      for (int a = 0; a < 4; ++a)
#pragma unroll
        for (int c = 0; c < 3; ++c) t += xs[16 + a * 3 + c] * sh1[c] * wk[168 + a * 8 + b];
      k0v[b] = (sh0 * s0 + INV3 * t) * NTP;
    }
#pragma unroll
    for (int b = 0; b < 2; ++b)
#pragma unroll
      for (int c = 0; c < 3; ++c) {
        float s0 = 0.f, t = 0.f;
#pragma unroll
        for (int a = 0; a < 16; ++a) s0 += xs[a] * wk[128 + a * 2 + b];
#pragma unroll
        for (int a = 0; a < 4; ++a) t += xs[16 + a * 3 + c] * wk[160 + a * 2 + b];
        k1v[b * 3 + c] = (sh1[c] * s0 + sh0 * t) * NTP;
      }
    float lg = 0.f;
#pragma unroll
    for (int i = 0; i < 8; ++i)
#pragma unroll
      for (int j = 0; j < 8; ++j) lg += qd[i] * k0v[j] * wd0s[i * 8 + j];
    float lt = 0.f;
#pragma unroll
    for (int i = 0; i < 2; ++i)
#pragma unroll
      for (int j = 0; j < 2; ++j)
#pragma unroll
        for (int c = 0; c < 3; ++c) lt += qd[8 + i * 3 + c] * k1v[j * 3 + c] * wd1s[i * 2 + j];
    lg = (lg + INV3 * lt) * NATT;
    if (ev && hh == 0) alog[e] = lg;

    // ================= V path: MLP -> tp_fc -> message ====================
    mlp_tile(xa, w1v, b1v, w2v, b2v, w3v, b3v, 25, 400, abuf, wbuf, lane);
    const float* wv = wbuf + m * 400;   // [w00 16x16 |256| w01 16x4 |320| w10 4x4 |336| w11 4x16]
    if (hh == 0) {
#pragma unroll
      for (int b = 0; b < 16; ++b) {
        float s0 = 0.f, t = 0.f;
#pragma unroll
        for (int a = 0; a < 16; ++a) s0 += xs[a] * wv[a * 16 + b];
#pragma unroll
        for (int a = 0; a < 4; ++a)
#pragma unroll
          for (int c = 0; c < 3; ++c) t += xs[16 + a * 3 + c] * sh1[c] * wv[336 + a * 16 + b];
        if (ev) vout[(size_t)e * 28 + b] = (sh0 * s0 + INV3 * t) * NTP;
      }
    } else {
#pragma unroll
      for (int b = 0; b < 4; ++b)
#pragma unroll
        for (int c = 0; c < 3; ++c) {
          float s0 = 0.f, t = 0.f;
#pragma unroll
          for (int a = 0; a < 16; ++a) s0 += xs[a] * wv[256 + a * 4 + b];
#pragma unroll
          for (int a = 0; a < 4; ++a) t += xs[16 + a * 3 + c] * wv[320 + a * 4 + b];
          if (ev) vout[(size_t)e * 28 + 16 + b * 3 + c] = (sh1[c] * s0 + sh0 * t) * NTP;
        }
    }
  }
}

// ------------------- Kernels 3-6: segment softmax + agg -------------------
static __device__ __forceinline__ void atomicMaxFloat(float* addr, float val) {
  if (val >= 0.f) atomicMax((int*)addr, __float_as_int(val));
  else            atomicMin((unsigned int*)addr, (unsigned int)__float_as_int(val));
}

__global__ void tpconv_init_kernel(float* amax, float* z, float* agg, int N) {
  int i = blockIdx.x * blockDim.x + threadIdx.x;
  if (i < N) { amax[i] = -__builtin_inff(); z[i] = 0.f; }
  if (i < N * 28) agg[i] = 0.f;
}

__global__ void tpconv_amax_kernel(const float* __restrict__ alog, const int* __restrict__ eidx,
                                   float* amax, int E) {
  int e = blockIdx.x * blockDim.x + threadIdx.x;
  if (e >= E) return;
  atomicMaxFloat(&amax[eidx[(size_t)E + e]], alog[e]);
}

__global__ void tpconv_p_kernel(const float* __restrict__ alog, const int* __restrict__ eidx,
                                const float* __restrict__ amax, float* p, float* z, int E) {
  int e = blockIdx.x * blockDim.x + threadIdx.x;
  if (e >= E) return;
  int d = eidx[(size_t)E + e];
  float v = expf(alog[e] - amax[d]);
  p[e] = v;
  atomicAdd(&z[d], v);
}

__global__ void tpconv_agg_kernel(const float* __restrict__ p, const float* __restrict__ vout,
                                  const int* __restrict__ eidx, const float* __restrict__ z,
                                  float* agg, long long total, int E) {
  long long i = (long long)blockIdx.x * blockDim.x + threadIdx.x;
  if (i >= total) return;
  int e = (int)(i / 28), c = (int)(i % 28);
  int d = eidx[(size_t)E + e];
  atomicAdd(&agg[(size_t)d * 28 + c], (p[e] / z[d]) * vout[(size_t)e * 28 + c]);
}

// ------------------- Kernel 7: output ir_linear + residual ----------------
__global__ void tpconv_out_kernel(const float* __restrict__ agg,
                                  const float* __restrict__ Wout0, const float* __restrict__ Wout1,
                                  const float* __restrict__ na, float* __restrict__ out, int N) {
  int nd = blockIdx.x * blockDim.x + threadIdx.x;
  if (nd >= N) return;
  const float* a = agg + (size_t)nd * 28;
  const float* r = na + (size_t)nd * 56;
  float* o = out + (size_t)nd * 56;
#pragma unroll
  for (int j = 0; j < 32; ++j) {
    float s = 0.f;
#pragma unroll
    for (int i = 0; i < 16; ++i) s += a[i] * Wout0[i * 32 + j];
    o[j] = s * 0.25f + r[j];            // 1/sqrt(16)
  }
#pragma unroll
  for (int j = 0; j < 8; ++j)
#pragma unroll
    for (int c = 0; c < 3; ++c) {
      float s = 0.f;
#pragma unroll
      for (int i = 0; i < 4; ++i) s += a[16 + i * 3 + c] * Wout1[i * 8 + j];
      o[32 + j * 3 + c] = s * 0.5f + r[32 + j * 3 + c];  // 1/sqrt(4)
    }
}

// --------------------------------- launch ---------------------------------
extern "C" void kernel_launch(void* const* d_in, const int* in_sizes, int n_in,
                              void* d_out, int out_size, void* d_ws, size_t ws_size,
                              hipStream_t stream) {
  (void)n_in; (void)out_size; (void)ws_size;
  const float* node_attr = (const float*)d_in[0];
  const float* edge_attr = (const float*)d_in[1];
  const float* edge_sh   = (const float*)d_in[2];
  const int*   edge_index= (const int*)d_in[3];
  const float* W_in0 = (const float*)d_in[4];
  const float* W_in1 = (const float*)d_in[5];
  const float* Wq0   = (const float*)d_in[6];
  const float* Wq1   = (const float*)d_in[7];
  const float* Wd0   = (const float*)d_in[8];
  const float* Wd1   = (const float*)d_in[9];
  const float* W_out0= (const float*)d_in[10];
  const float* W_out1= (const float*)d_in[11];
  const float* fcW1 = (const float*)d_in[12]; const float* fcb1 = (const float*)d_in[13];
  const float* fcW2 = (const float*)d_in[14]; const float* fcb2 = (const float*)d_in[15];
  const float* fcW3 = (const float*)d_in[16]; const float* fcb3 = (const float*)d_in[17];
  const float* fkW1 = (const float*)d_in[18]; const float* fkb1 = (const float*)d_in[19];
  const float* fkW2 = (const float*)d_in[20]; const float* fkb2 = (const float*)d_in[21];
  const float* fkW3 = (const float*)d_in[22]; const float* fkb3 = (const float*)d_in[23];

  const int N = in_sizes[0] / 56;
  const int E = in_sizes[1] / 32;

  float* ws   = (float*)d_ws;
  float* xw   = ws;                        // N*28
  float* qw   = xw   + (size_t)N * 28;     // N*14
  float* amax = qw   + (size_t)N * 14;     // N
  float* zsum = amax + N;                  // N
  float* agg  = zsum + N;                  // N*28
  float* alog = agg  + (size_t)N * 28;     // E
  float* pbuf = alog + E;                  // E
  float* vbuf = pbuf + E;                  // E*28
  float* outp = (float*)d_out;

  tpconv_node_kernel<<<(N + 127) / 128, 128, 0, stream>>>(
      node_attr, W_in0, W_in1, Wq0, Wq1, xw, qw, N);

  int ntiles = (E + 15) / 16;
  int eblocks = (ntiles + 3) / 4;
  if (eblocks > 512) eblocks = 512;
  tpconv_edge_kernel<<<eblocks, 128, L_TOTAL, stream>>>(
      edge_attr, edge_sh, edge_index, xw, qw,
      fkW1, fkb1, fkW2, fkb2, fkW3, fkb3,
      fcW1, fcb1, fcW2, fcb2, fcW3, fcb3,
      Wd0, Wd1, alog, vbuf, E);

  tpconv_init_kernel<<<(N * 28 + 255) / 256, 256, 0, stream>>>(amax, zsum, agg, N);
  tpconv_amax_kernel<<<(E + 255) / 256, 256, 0, stream>>>(alog, edge_index, amax, E);
  tpconv_p_kernel<<<(E + 255) / 256, 256, 0, stream>>>(alog, edge_index, amax, pbuf, zsum, E);
  long long total = (long long)E * 28;
  tpconv_agg_kernel<<<(unsigned)((total + 255) / 256), 256, 0, stream>>>(
      pbuf, vbuf, edge_index, zsum, agg, total, E);
  tpconv_out_kernel<<<(N + 127) / 128, 128, 0, stream>>>(
      agg, W_out0, W_out1, node_attr, outp, N);
}
